// QLoRALinear_62594853372313
// MI455X (gfx1250) — matrix operand order repaired and checked by reading, not compiled, not run on hardware
//
#include <hip/hip_runtime.h>
#include <hip/hip_bf16.h>

typedef __attribute__((ext_vector_type(16))) _Float16 v16h;
typedef __attribute__((ext_vector_type(8)))  _Float16 v8h;
typedef __attribute__((ext_vector_type(8)))  float    v8f;
typedef __attribute__((ext_vector_type(4)))  float    v4f;

__constant__ float NF4_TAB[16] = {
    -1.0f, -0.6961928009986877f, -0.5250730514526367f, -0.39491748809814453f,
    -0.28444138169288635f, -0.18477343022823334f, -0.09105003625154495f, 0.0f,
    0.07958029955625534f, 0.16093020141124725f, 0.24611230194568634f,
    0.33791524171829224f, 0.44070982933044434f, 0.5626170039176941f,
    0.7229568362236023f, 1.0f};

static __device__ __forceinline__ v16h cat8(v8h lo, v8h hi) {
  return __builtin_shufflevector(lo, hi, 0, 1, 2, 3, 4, 5, 6, 7,
                                 8, 9, 10, 11, 12, 13, 14, 15);
}

// ---------------------------------------------------------------------------
// Pass 1: x (fp32) -> xh (f16). 8 elements / thread, b128 in / b128 out.
// ---------------------------------------------------------------------------
__global__ __launch_bounds__(256) void cvt_x_f16(const float* __restrict__ x,
                                                 _Float16* __restrict__ xh,
                                                 long long n) {
  long long i = ((long long)blockIdx.x * blockDim.x + threadIdx.x) * 8;
  if (i + 8 <= n) {
    v4f a = *(const v4f*)(x + i);
    v4f b = *(const v4f*)(x + i + 4);
    v8h o;
    o[0] = (_Float16)a[0]; o[1] = (_Float16)a[1];
    o[2] = (_Float16)a[2]; o[3] = (_Float16)a[3];
    o[4] = (_Float16)b[0]; o[5] = (_Float16)b[1];
    o[6] = (_Float16)b[2]; o[7] = (_Float16)b[3];
    *(v8h*)(xh + i) = o;
  }
}

// ---------------------------------------------------------------------------
// Pass 2: W_eff[n,k] = NF4[q]*absmax + (alpha/rank) * up[n,:] . down[:,k]
// One block per output row; scaled up-row and NF4 table staged in LDS.
// ---------------------------------------------------------------------------
__global__ __launch_bounds__(256) void build_w_eff(
    const int* __restrict__ q, const float* __restrict__ absmax,
    const float* __restrict__ down, const float* __restrict__ up,
    const float* __restrict__ alpha, _Float16* __restrict__ wh,
    int in_f, int rank) {
  const int o = blockIdx.x;
  const int nblk = in_f >> 6;  // 64-element quant blocks per row
  __shared__ float s_nf4[16];
  __shared__ float s_up[32];
  const float scale = alpha[0] / (float)rank;
  if (threadIdx.x < 16) s_nf4[threadIdx.x] = NF4_TAB[threadIdx.x];
  if (threadIdx.x < (unsigned)rank)
    s_up[threadIdx.x] = up[(size_t)o * rank + threadIdx.x] * scale;
  __syncthreads();

  for (int i = threadIdx.x; i < in_f; i += blockDim.x) {
    float w = s_nf4[q[(size_t)o * in_f + i] & 15] *
              absmax[(size_t)o * nblk + (i >> 6)];
    float lora = 0.0f;
    for (int r = 0; r < rank; ++r)
      lora += s_up[r] * down[(size_t)r * in_f + i];
    wh[(size_t)o * in_f + i] = (_Float16)(w + lora);
  }
}

// ---------------------------------------------------------------------------
// Pass 3: out[M,N] = Xh[M,K] * Wh[N,K]^T + bias   via v_wmma_f32_16x16x32_f16
// Wave tile: 64M x 64N (4 A-tiles x 4 B-tiles, 16 WMMAs / 32-K step).
// B operand is double-buffered across K-phases (prefetched one full phase =
// 16 WMMAs ahead), A is loaded at phase top. sched_group_barrier pins each
// phase to <16 VMEM reads><16 WMMAs> so the scheduler cannot sink the
// prefetch back down and regalloc must keep both B banks live.
// Block: 8 waves (2Mw x 4Nw) -> 128M x 256N.
// ---------------------------------------------------------------------------

// One K-phase: consume (BLc,BHc) at k=KA, prefetch (BLn,BHn) at k=KB.
#define GEMM_PHASE(KA, BLc, BHc, KB, BLn, BHn)                                 \
  {                                                                            \
    v8h alo[4], ahi[4];                                                        \
    _Pragma("unroll") for (int mi = 0; mi < 4; ++mi) {                         \
      alo[mi] = *(const v8h*)(pa[mi] + (KA));                                  \
      ahi[mi] = *(const v8h*)(pa[mi] + (KA) + 8);                              \
    }                                                                          \
    _Pragma("unroll") for (int ti = 0; ti < 4; ++ti) {                         \
      BLn[ti] = *(const v8h*)(pb[ti] + (KB));                                  \
      BHn[ti] = *(const v8h*)(pb[ti] + (KB) + 8);                              \
    }                                                                          \
    v16h a[4];                                                                 \
    _Pragma("unroll") for (int mi = 0; mi < 4; ++mi)                           \
        a[mi] = cat8(alo[mi], ahi[mi]);                                        \
    _Pragma("unroll") for (int ti = 0; ti < 4; ++ti) {                         \
      v16h b = cat8(BLc[ti], BHc[ti]);                                         \
      _Pragma("unroll") for (int mi = 0; mi < 4; ++mi)                         \
          acc[mi][ti] = __builtin_amdgcn_wmma_f32_16x16x32_f16(                \
              false, a[mi], false, b, (short)0, acc[mi][ti], false, false);    \
    }                                                                          \
    __builtin_amdgcn_sched_group_barrier(0x0020, 16, 0); /* 16 VMEM read */    \
    __builtin_amdgcn_sched_group_barrier(0x0008, 16, 0); /* 16 WMMA      */    \
  }

__global__ __launch_bounds__(256) void qlora_wmma_gemm(
    const _Float16* __restrict__ X, const _Float16* __restrict__ W,
    const float* __restrict__ bias, float* __restrict__ out,
    int M, int N, int K) {
  const int lane = threadIdx.x & 31;
  const int wave = threadIdx.x >> 5;
  const int wm = wave >> 2;  // 0..1
  const int wn = wave & 3;   // 0..3
  const int m0 = blockIdx.y * 128 + wm * 64;
  const int n0 = blockIdx.x * 256 + wn * 64;

  const int lr = lane & 15;
  const int ah = (lane >> 4) * 16;  // A-matrix lane K offset
  const int bo = (lane >> 4) * 16;  // B-matrix lane K offset

  const _Float16* pa[4];
  const _Float16* pb[4];
#pragma unroll
  for (int i = 0; i < 4; ++i) {
    pa[i] = X + (size_t)(m0 + i * 16 + lr) * K + ah;
    pb[i] = W + (size_t)(n0 + i * 16 + lr) * K + bo;
  }

  v8f acc[4][4];
#pragma unroll
  for (int mi = 0; mi < 4; ++mi)
#pragma unroll
    for (int ti = 0; ti < 4; ++ti) acc[mi][ti] = (v8f){};

  // Prologue: fill buffer 0 with B tiles for k = 0.
  v8h b0l[4], b0h[4], b1l[4], b1h[4];
#pragma unroll
  for (int ti = 0; ti < 4; ++ti) {
    b0l[ti] = *(const v8h*)(pb[ti]);
    b0h[ti] = *(const v8h*)(pb[ti] + 8);
  }

  for (int k0 = 0; k0 < K; k0 += 64) {
    GEMM_PHASE(k0, b0l, b0h, k0 + 32, b1l, b1h);
    const int kn = (k0 + 64 < K) ? (k0 + 64) : 0;  // clamp dead tail prefetch
    GEMM_PHASE(k0 + 32, b1l, b1h, kn, b0l, b0h);
  }

  // Epilogue: C/D layout — lane holds N = lane&15, VGPR r holds M = r + 8*(lane>>4)
  const int nl = lr;
  const int mh = (lane >> 4) * 8;
#pragma unroll
  for (int ti = 0; ti < 4; ++ti) {
    const int ncol = n0 + ti * 16 + nl;
    const float bv = bias[ncol];
#pragma unroll
    for (int mi = 0; mi < 4; ++mi) {
      const size_t base = (size_t)(m0 + mi * 16 + mh) * N + ncol;
#pragma unroll
      for (int r = 0; r < 8; ++r)
        out[base + (size_t)r * N] = acc[mi][ti][r] + bv;
    }
  }
}

// ---------------------------------------------------------------------------
extern "C" void kernel_launch(void* const* d_in, const int* in_sizes, int n_in,
                              void* d_out, int out_size, void* d_ws,
                              size_t ws_size, hipStream_t stream) {
  (void)n_in; (void)out_size; (void)ws_size;
  const float* x      = (const float*)d_in[0];
  const float* absmax = (const float*)d_in[1];
  const float* bias   = (const float*)d_in[2];
  const float* down   = (const float*)d_in[3];
  const float* up     = (const float*)d_in[4];
  const float* alpha  = (const float*)d_in[5];
  const int*   q      = (const int*)d_in[6];
  float* out = (float*)d_out;

  // Derive dims from input sizes: bias=[out_f], up=[out_f,rank],
  // down=[rank,in_f], x=[n_tok,in_f]
  const int out_f = in_sizes[2];
  const int rank  = in_sizes[4] / out_f;
  const int in_f  = in_sizes[3] / rank;
  const int n_tok = in_sizes[0] / in_f;

  _Float16* xh = (_Float16*)d_ws;                     // [n_tok, in_f]
  _Float16* wh = xh + (size_t)n_tok * in_f;           // [out_f, in_f]

  // Pass 1: x -> f16
  {
    long long n = (long long)n_tok * in_f;
    long long threads = n / 8;
    int blocks = (int)((threads + 255) / 256);
    cvt_x_f16<<<blocks, 256, 0, stream>>>(x, xh, n);
  }
  // Pass 2: fold NF4 dequant + LoRA into W_eff (f16)
  build_w_eff<<<out_f, 256, 0, stream>>>(q, absmax, down, up, alpha, wh,
                                         in_f, rank);
  // Pass 3: WMMA GEMM + bias
  {
    dim3 grid(out_f / 256, n_tok / 128);
    qlora_wmma_gemm<<<grid, 256, 0, stream>>>(xh, wh, bias, out,
                                              n_tok, out_f, in_f);
  }
}
